// VisionTransformer_74096775790770
// MI455X (gfx1250) — compile-verified, matrix operand order checked by reading, not run on hardware
//
#include <hip/hip_runtime.h>
#include <stdint.h>

// ---------------------------------------------------------------------------
// Types / WMMA fragments (CDNA5 gfx1250, wave32)
// ---------------------------------------------------------------------------
typedef __bf16 bf16_t;
typedef __attribute__((ext_vector_type(16))) __bf16 v16bf;
typedef __attribute__((ext_vector_type(8)))  float  v8f;

#define D_MODEL 768
#define D_FF    3072
#define N_HEADS 12
#define HEAD_DIM 64
#define BATCH   4

union U32B { uint32_t u; __bf16 h[2]; };

// K index of pair q within a 16-bit A/B fragment (16x32 / 32x16), before the
// +8 per-lane-group offset.  (CDNA5 ISA "16-bit A-Matrix 16x32")
__device__ __forceinline__ int pairK(int q) { return q < 4 ? 2 * q : 16 + 2 * (q - 4); }

// Load a v16bf fragment where lane holds a fixed row/col and K varies with the
// documented pair pattern; p points at (row base + k0), g = laneId>>4.
// The pair pattern is two contiguous 16B runs -> compiler emits b128 loads.
__device__ __forceinline__ v16bf frag_load_pairs(const bf16_t* p, int g) {
  v16bf f;
#pragma unroll
  for (int q = 0; q < 8; ++q) {
    U32B t;
    t.u = *reinterpret_cast<const uint32_t*>(p + pairK(q) + g * 8);
    f[2 * q]     = t.h[0];
    f[2 * q + 1] = t.h[1];
  }
  return f;
}

__device__ __forceinline__ float wred_max16(float v) {
#pragma unroll
  for (int m = 1; m < 16; m <<= 1) v = fmaxf(v, __shfl_xor(v, m, 32));
  return v;
}
__device__ __forceinline__ float wred_sum16(float v) {
#pragma unroll
  for (int m = 1; m < 16; m <<= 1) v += __shfl_xor(v, m, 32);
  return v;
}

__device__ __forceinline__ float gelu_tanh(float x) {
  float x3 = x * x * x;
  return 0.5f * x * (1.0f + tanhf(0.7978845608028654f * (x + 0.044715f * x3)));
}

__device__ __forceinline__ v8f wmma_bf16(v16bf a, v16bf b, v8f c) {
  return __builtin_amdgcn_wmma_f32_16x16x32_bf16(false, a, false, b, (short)0, c,
                                                 false, false);
}

// ---------------------------------------------------------------------------
// Weight convert: fp32 [K][N] -> bf16 [N][K] (optionally no transpose)
// ---------------------------------------------------------------------------
__global__ void convert_w_k(const float* __restrict__ W, bf16_t* __restrict__ Wt,
                            int K, int N, int transpose) {
  long idx = (long)blockIdx.x * blockDim.x + threadIdx.x;
  long total = (long)K * N;
  if (idx >= total) return;
  int n = (int)(idx / K), k = (int)(idx % K);
  float v = transpose ? W[(long)k * N + n] : W[idx];
  Wt[idx] = (bf16_t)v;
}

// ---------------------------------------------------------------------------
// Quadtree schedule (replicates numpy bookkeeping). Single thread; tiny work.
// meta: [0..3]=token count per layer, [4..6]=old+keep per split layer,
//       [8..10]=4*nsplit per split layer
// ---------------------------------------------------------------------------
__global__ void schedule_k(const int* ol0, const int* ol1, const int* ol2,
                           int* meta, int* scaleA, int* idxA, int* scaleB, int* idxB,
                           int* map_src, int* ssrc, int* fscale, int* fidx, int Npad) {
  if (threadIdx.x != 0 || blockIdx.x != 0) return;
  const int* ols[3] = {ol0, ol1, ol2};
  int* cs = scaleA; int* ci = idxA; int* ns = scaleB; int* ni = idxB;
  int N = 256;
  for (int i = 0; i < 256; ++i) { cs[i] = 0; ci[i] = i; }
  meta[0] = 256;
  int ps = 16;
  for (int blk = 0; blk < 3; ++blk) {
    int* msrc = map_src + blk * Npad;
    int* ss   = ssrc    + blk * Npad;
    int no = 0, nk = 0, nsp = 0;
    for (int p = 0; p < N; ++p)
      if (cs[p] != blk) { msrc[no] = p; ns[no] = cs[p]; ni[no] = ci[p]; ++no; }
    for (int p = 0; p < N; ++p)
      if (cs[p] == blk && ols[blk][ci[p]] == 1) {
        msrc[no + nk] = p; ns[no + nk] = cs[p]; ni[no + nk] = ci[p]; ++nk;
      }
    for (int p = 0; p < N; ++p)
      if (cs[p] == blk && ols[blk][ci[p]] == 0) {
        ss[nsp] = p;
        int r = ci[p] / ps, c = ci[p] % ps;
        int base = no + nk + nsp * 4;
        int rr[4] = {2 * r, 2 * r, 2 * r + 1, 2 * r + 1};
        int cc[4] = {2 * c, 2 * c + 1, 2 * c, 2 * c + 1};
        for (int ch = 0; ch < 4; ++ch) {
          ns[base + ch] = blk + 1;
          ni[base + ch] = rr[ch] * (2 * ps) + cc[ch];
        }
        ++nsp;
      }
    meta[4 + blk] = no + nk;
    meta[8 + blk] = nsp * 4;
    N = no + nk + nsp * 4;
    meta[1 + blk] = N;
    int* t;
    t = cs; cs = ns; ns = t;
    t = ci; ci = ni; ni = t;
    ps *= 2;
  }
  for (int p = 0; p < N; ++p) { fscale[p] = cs[p]; fidx[p] = ci[p]; }
}

// ---------------------------------------------------------------------------
// Patch embed: X[b, p, :] = patch(b,p) @ conv_w^T + conv_b + pos.
// conv_w is (D, C*PS*PS) = [n][k] already, pre-converted to bf16.
// grid (16, 12, B), block 128 (4 waves, 16 n-cols each).
// ---------------------------------------------------------------------------
__global__ __launch_bounds__(128)
void patch_embed_k(const float* __restrict__ im, const bf16_t* __restrict__ WconvT,
                   const float* __restrict__ convb, const float* __restrict__ pos,
                   float* __restrict__ X, int Npad) {
  __shared__ bf16_t As[16][32];
  const int b = blockIdx.z;
  const int mbase = blockIdx.x * 16;
  const int wave = threadIdx.x >> 5, lane = threadIdx.x & 31;
  const int g = lane >> 4, mrow = lane & 15;
  const int nCol = blockIdx.y * 64 + wave * 16 + mrow;
  v8f acc = {};
  for (int k0 = 0; k0 < D_MODEL; k0 += 32) {
    for (int t = threadIdx.x; t < 512; t += 128) {
      int m = t >> 5, kk = t & 31, k = k0 + kk;
      int patch = mbase + m;
      int pr = patch >> 4, pc = patch & 15;
      int c = k >> 8, rem = k & 255, ph = rem >> 4, pw = rem & 15;
      float v = im[(((long)b * 3 + c) * 256 + (pr * 16 + ph)) * 256 + (pc * 16 + pw)];
      As[m][kk] = (bf16_t)v;
    }
    __syncthreads();
    v16bf a = frag_load_pairs(&As[mrow][0], g);
    v16bf w = frag_load_pairs(WconvT + (long)nCol * D_MODEL + k0, g);
    acc = wmma_bf16(a, w, acc);
    __syncthreads();
  }
  float bb = convb[nCol];
#pragma unroll
  for (int r = 0; r < 8; ++r) {
    int row = mbase + r + 8 * g;
    X[((long)b * Npad + row) * D_MODEL + nCol] = acc[r] + bb + pos[(long)row * D_MODEL + nCol];
  }
}

// ---------------------------------------------------------------------------
// LayerNorm: fp32 row -> normalized bf16 row.  One wave per token row.
// ---------------------------------------------------------------------------
__global__ __launch_bounds__(256)
void layernorm_k(const float* __restrict__ X, const float* __restrict__ s,
                 const float* __restrict__ bta, bf16_t* __restrict__ H,
                 const int* __restrict__ cntPtr, long batchStride) {
  const int wave = threadIdx.x >> 5, lane = threadIdx.x & 31;
  const int row = blockIdx.x * 8 + wave;
  const int cnt = cntPtr[0];
  if (row >= cnt) return;
  const float* xr = X + (long)blockIdx.z * batchStride + (long)row * D_MODEL;
  float sum = 0.f, sq = 0.f;
  for (int i = lane; i < D_MODEL; i += 32) { float v = xr[i]; sum += v; sq += v * v; }
#pragma unroll
  for (int m = 1; m < 32; m <<= 1) { sum += __shfl_xor(sum, m, 32); sq += __shfl_xor(sq, m, 32); }
  float mean = sum * (1.0f / D_MODEL);
  float var  = sq * (1.0f / D_MODEL) - mean * mean;
  float rstd = rsqrtf(var + 1e-5f);
  bf16_t* hr = H + (long)blockIdx.z * batchStride + (long)row * D_MODEL;
  for (int i = lane; i < D_MODEL; i += 32)
    hr[i] = (bf16_t)((xr[i] - mean) * rstd * s[i] + bta[i]);
}

// ---------------------------------------------------------------------------
// Generic WMMA GEMM:  out[M,N] = A[M,K](bf16) @ Wt^T + bias, epilogue by MODE.
//  MODE 0: bf16 out          MODE 1: gelu -> bf16 out
//  MODE 2: fp32 out = resid + acc + bias
//  MODE 3: fp32 out at row offset dstBase[0] (split children)
// grid (Mtiles, N/128, B), block 128 (4 waves, each wave owns TWO 16-wide
// column tiles sharing one A fragment).  Double-buffered LDS A staging ->
// one barrier per K step; weight stream prefetched two steps ahead
// (global_prefetch_b8; weights stay L2-resident on the 192MB L2).
// ---------------------------------------------------------------------------
template <int MODE>
__global__ __launch_bounds__(128)
void gemm16_k(const bf16_t* __restrict__ A, const bf16_t* __restrict__ Wt,
              const float* __restrict__ bias, float* __restrict__ outF,
              bf16_t* __restrict__ outB, const float* __restrict__ resid,
              const int* __restrict__ cntPtr, const int* __restrict__ dstBasePtr,
              int K, long aBatch, long oBatch, long rBatch, int aStride, int oStride) {
  const int cnt = cntPtr[0];
  const int mbase = blockIdx.x * 16;
  if (mbase >= cnt) return;
  const int wave = threadIdx.x >> 5, lane = threadIdx.x & 31;
  const int g = lane >> 4, mrow = lane & 15;
  const int nBase = blockIdx.y * 128 + wave * 32;
  const int n0 = nBase + mrow, n1 = nBase + 16 + mrow;
  const bf16_t* Ab = A + (long)blockIdx.z * aBatch + (long)mbase * aStride;
  const bf16_t* W0 = Wt + (long)n0 * K;
  const bf16_t* W1 = Wt + (long)n1 * K;
  __shared__ bf16_t As[2][16][32];

  auto stage = [&](int buf, int k0) {
    for (int t = threadIdx.x; t < 256; t += 128) {  // 256 dwords = 16x32 bf16
      int m = t >> 4, kk = (t & 15) * 2;
      *reinterpret_cast<uint32_t*>(&As[buf][m][kk]) =
          *reinterpret_cast<const uint32_t*>(Ab + (long)m * aStride + k0 + kk);
    }
  };

  stage(0, 0);
  v8f acc0 = {}, acc1 = {};
  int buf = 0;
  for (int k0 = 0; k0 < K; k0 += 32) {
    __syncthreads();  // staging of As[buf] complete; reads of As[buf^1] done
    if (k0 + 32 < K) stage(buf ^ 1, k0 + 32);
    if (k0 + 64 < K) {
      __builtin_prefetch(W0 + k0 + 64, 0, 3);
      __builtin_prefetch(W1 + k0 + 64, 0, 3);
    }
    v16bf a  = frag_load_pairs(&As[buf][mrow][0], g);
    v16bf w0 = frag_load_pairs(W0 + k0, g);
    v16bf w1 = frag_load_pairs(W1 + k0, g);
    acc0 = wmma_bf16(a, w0, acc0);
    acc1 = wmma_bf16(a, w1, acc1);
    buf ^= 1;
  }

  const float bb0 = bias ? bias[n0] : 0.0f;
  const float bb1 = bias ? bias[n1] : 0.0f;
#pragma unroll
  for (int r = 0; r < 8; ++r) {
    int row = mbase + r + 8 * g;
    float v0 = acc0[r] + bb0;
    float v1 = acc1[r] + bb1;
    if constexpr (MODE == 0) {
      long o = (long)blockIdx.z * oBatch + (long)row * oStride;
      outB[o + n0] = (bf16_t)v0;
      outB[o + n1] = (bf16_t)v1;
    } else if constexpr (MODE == 1) {
      long o = (long)blockIdx.z * oBatch + (long)row * oStride;
      outB[o + n0] = (bf16_t)gelu_tanh(v0);
      outB[o + n1] = (bf16_t)gelu_tanh(v1);
    } else if constexpr (MODE == 2) {
      long o = (long)blockIdx.z * oBatch + (long)row * oStride;
      long rb = (long)blockIdx.z * rBatch + (long)row * oStride;
      outF[o + n0] = resid[rb + n0] + v0;
      outF[o + n1] = resid[rb + n1] + v1;
    } else {  // MODE == 3
      int drow = dstBasePtr[0] + row;
      long o = (long)blockIdx.z * oBatch + (long)drow * oStride;
      outF[o + n0] = v0;
      outF[o + n1] = v1;
    }
  }
}

// ---------------------------------------------------------------------------
// Flash-style attention.  One wave per (16-query tile, head, batch).
// QKV layout: [b][token][2304] with Q|K|V slabs of 768; head h at feat h*64.
// ---------------------------------------------------------------------------
__global__ __launch_bounds__(32)
void attention_k(const bf16_t* __restrict__ QKV, bf16_t* __restrict__ O,
                 const int* __restrict__ cntPtr, int Npad) {
  const int lane = threadIdx.x & 31;
  const int qbase = blockIdx.x * 16;
  const int h = blockIdx.y, b = blockIdx.z;
  const int cnt = cntPtr[0];
  if (qbase >= cnt) return;
  const int g = lane >> 4, m = lane & 15;
  const bf16_t* base = QKV + (long)b * Npad * 2304;

  int qrow = qbase + m; if (qrow > cnt - 1) qrow = cnt - 1;
  const bf16_t* qp = base + (long)qrow * 2304 + h * HEAD_DIM;
  v16bf qf0 = frag_load_pairs(qp, g);
  v16bf qf1 = frag_load_pairs(qp + 32, g);

  float rowmax[8], rowsum[8];
#pragma unroll
  for (int r = 0; r < 8; ++r) { rowmax[r] = -3.0e38f; rowsum[r] = 0.0f; }
  v8f o0 = {}, o1 = {}, o2 = {}, o3 = {};

  __shared__ bf16_t P[16][32];
  __shared__ bf16_t Vt[32][64];

  for (int k0 = 0; k0 < cnt; k0 += 32) {
    v8f sacc[2] = {{}, {}};
#pragma unroll
    for (int kt = 0; kt < 2; ++kt) {
      int key = k0 + kt * 16 + m;
      int keyc = key > cnt - 1 ? cnt - 1 : key;
      const bf16_t* kp = base + (long)keyc * 2304 + 768 + h * HEAD_DIM;
      v16bf kf0 = frag_load_pairs(kp, g);
      v16bf kf1 = frag_load_pairs(kp + 32, g);
      sacc[kt] = wmma_bf16(qf0, kf0, sacc[kt]);
      sacc[kt] = wmma_bf16(qf1, kf1, sacc[kt]);
      bool oob = key >= cnt;
#pragma unroll
      for (int r = 0; r < 8; ++r)
        sacc[kt][r] = oob ? -3.0e38f : sacc[kt][r] * 0.125f;  // HD^-0.5 = 1/8
    }
    // online softmax bookkeeping (rows m = r + 8g live in the same lane group)
    float pl[2][8];
#pragma unroll
    for (int r = 0; r < 8; ++r) {
      float vmax = wred_max16(fmaxf(sacc[0][r], sacc[1][r]));
      float nm = fmaxf(rowmax[r], vmax);
      float scale = expf(rowmax[r] - nm);
      rowmax[r] = nm;
      rowsum[r] *= scale;
      o0[r] *= scale; o1[r] *= scale; o2[r] *= scale; o3[r] *= scale;
      float p0 = expf(sacc[0][r] - nm);
      float p1 = expf(sacc[1][r] - nm);
      pl[0][r] = p0; pl[1][r] = p1;
      rowsum[r] += wred_sum16(p0 + p1);
    }
    // stage P as [query][key] (A-fragment layout source) and the V tile in LDS
#pragma unroll
    for (int kt = 0; kt < 2; ++kt)
#pragma unroll
      for (int r = 0; r < 8; ++r)
        P[r + 8 * g][kt * 16 + m] = (bf16_t)pl[kt][r];
    for (int t = lane; t < 1024; t += 32) {  // 32 keys x 64 feats as dwords
      int kk = t >> 5, f2 = (t & 31) * 2;
      int key = k0 + kk; if (key > cnt - 1) key = cnt - 1;
      *reinterpret_cast<uint32_t*>(&Vt[kk][f2]) = *reinterpret_cast<const uint32_t*>(
          base + (long)key * 2304 + 1536 + h * HEAD_DIM + f2);
    }
    __syncthreads();
    v16bf pa = frag_load_pairs(&P[m][0], g);
#pragma unroll
    for (int ft = 0; ft < 4; ++ft) {
      v16bf vb;
#pragma unroll
      for (int e = 0; e < 16; ++e) {
        int k = pairK(e >> 1) + g * 8 + (e & 1);
        vb[e] = Vt[k][ft * 16 + m];
      }
      v8f* op = ft == 0 ? &o0 : ft == 1 ? &o1 : ft == 2 ? &o2 : &o3;
      *op = wmma_bf16(pa, vb, *op);
    }
    __syncthreads();
  }
#pragma unroll
  for (int r = 0; r < 8; ++r) {
    int row = qbase + r + 8 * g;
    if (row < cnt) {
      float inv = 1.0f / rowsum[r];
      long ob = ((long)b * Npad + row) * D_MODEL + h * HEAD_DIM + m;
      O[ob +  0] = (bf16_t)(o0[r] * inv);
      O[ob + 16] = (bf16_t)(o1[r] * inv);
      O[ob + 32] = (bf16_t)(o2[r] * inv);
      O[ob + 48] = (bf16_t)(o3[r] * inv);
    }
  }
}

// ---------------------------------------------------------------------------
// Token split helpers
// ---------------------------------------------------------------------------
__global__ __launch_bounds__(256)
void copy_oldkeep_k(const float* __restrict__ Xs, float* __restrict__ Xd,
                    const int* __restrict__ msrc, const int* __restrict__ okPtr, int Npad) {
  const int ok = okPtr[0];
  const int row = blockIdx.x;
  if (row >= ok) return;
  const int b = blockIdx.z, src = msrc[row];
  for (int t = threadIdx.x; t < D_MODEL; t += 256)
    Xd[((long)b * Npad + row) * D_MODEL + t] = Xs[((long)b * Npad + src) * D_MODEL + t];
}

__global__ __launch_bounds__(64)
void gather_split_k(const float* __restrict__ Xs, bf16_t* __restrict__ Sg,
                    const int* __restrict__ ssrc, const int* __restrict__ s4Ptr, int Npad) {
  const int n4 = s4Ptr[0];
  const int row = blockIdx.x;
  if (row >= n4) return;
  const int b = blockIdx.z, src = ssrc[row >> 2], ch = row & 3;
  for (int t = threadIdx.x; t < 192; t += 64)
    Sg[((long)b * Npad + row) * 192 + t] =
        (bf16_t)Xs[((long)b * Npad + src) * D_MODEL + ch * 192 + t];
}

__global__ __launch_bounds__(256)
void write_out_k(const float* __restrict__ X, const int* __restrict__ fscale,
                 const int* __restrict__ fidx, float* __restrict__ out, int Nf, int Npad) {
  const int row = blockIdx.x, b = blockIdx.z;
  for (int t = threadIdx.x; t < D_MODEL; t += 256)
    out[((long)b * Nf + row) * D_MODEL + t] = X[((long)b * Npad + row) * D_MODEL + t];
  if (b == 0 && threadIdx.x == 0) {
    long cbase = (long)BATCH * Nf * D_MODEL;
    out[cbase + (long)row * 2 + 0] = (float)fscale[row];
    out[cbase + (long)row * 2 + 1] = (float)fidx[row];
  }
}

// ---------------------------------------------------------------------------
// Host driver
// ---------------------------------------------------------------------------
extern "C" void kernel_launch(void* const* d_in, const int* in_sizes, int n_in,
                              void* d_out, int out_size, void* d_ws, size_t ws_size,
                              hipStream_t stream) {
  (void)in_sizes; (void)n_in; (void)ws_size;
  // final token count from output size: out = B*Nf*D (x) + Nf*2 (coords)
  int Nf = out_size / (BATCH * D_MODEL + 2);
  if (Nf < 256) Nf = 256;
  const int Npad = (Nf + 15) & ~15;

  const float* im     = (const float*)d_in[0];
  const int*   ol0    = (const int*)d_in[1];
  const int*   ol1    = (const int*)d_in[2];
  const int*   ol2    = (const int*)d_in[3];
  const float* conv_w = (const float*)d_in[4];
  const float* conv_b = (const float*)d_in[5];
  const float* pos    = (const float*)d_in[6];
  auto blk = [&](int l, int j) { return (const float*)d_in[7 + 12 * l + j]; };
  auto spl = [&](int l, int j) { return (const float*)d_in[7 + 48 + 2 * l + j]; };
  // block param order: 0 ln1_s, 1 ln1_b, 2 wqkv, 3 bqkv, 4 wproj, 5 bproj,
  //                    6 ln2_s, 7 ln2_b, 8 w1, 9 b1, 10 w2, 11 b2

  // workspace carve (256B aligned)
  char* wsp = (char*)d_ws;
  auto carve = [&](size_t bytes) -> char* {
    char* p = wsp;
    wsp += (bytes + 255) & ~(size_t)255;
    return p;
  };
  int* meta    = (int*)carve(16 * sizeof(int));
  int* scaleA  = (int*)carve((size_t)Npad * sizeof(int));
  int* idxA    = (int*)carve((size_t)Npad * sizeof(int));
  int* scaleB  = (int*)carve((size_t)Npad * sizeof(int));
  int* idxB    = (int*)carve((size_t)Npad * sizeof(int));
  int* map_src = (int*)carve((size_t)3 * Npad * sizeof(int));
  int* ssrc    = (int*)carve((size_t)3 * Npad * sizeof(int));
  int* fscale  = (int*)carve((size_t)Npad * sizeof(int));
  int* fidx    = (int*)carve((size_t)Npad * sizeof(int));

  bf16_t* wconvT = (bf16_t*)carve((size_t)768 * 768 * 2);
  bf16_t *wqkvT[4], *wprojT[4], *w1T[4], *w2T[4], *wsplT[3];
  for (int l = 0; l < 4; ++l) {
    wqkvT[l]  = (bf16_t*)carve((size_t)2304 * 768 * 2);
    wprojT[l] = (bf16_t*)carve((size_t)768 * 768 * 2);
    w1T[l]    = (bf16_t*)carve((size_t)3072 * 768 * 2);
    w2T[l]    = (bf16_t*)carve((size_t)768 * 3072 * 2);
  }
  for (int l = 0; l < 3; ++l) wsplT[l] = (bf16_t*)carve((size_t)768 * 192 * 2);

  const long NPD  = (long)Npad * D_MODEL;
  const long NPQ  = (long)Npad * 2304;
  const long NPF  = (long)Npad * D_FF;
  float*  X0  = (float*)carve((size_t)BATCH * NPD * 4);
  float*  X1  = (float*)carve((size_t)BATCH * NPD * 4);
  bf16_t* Hbf = (bf16_t*)carve((size_t)BATCH * NPD * 2);
  bf16_t* QKV = (bf16_t*)carve((size_t)BATCH * NPQ * 2);
  bf16_t* Obf = (bf16_t*)carve((size_t)BATCH * NPD * 2);
  bf16_t* Gbf = (bf16_t*)carve((size_t)BATCH * NPF * 2);
  bf16_t* Sg  = (bf16_t*)carve((size_t)BATCH * Npad * 192 * 2);

  // weight conversion (bf16, transposed to [N][K]; stays L2-resident)
  auto cw = [&](const float* src, bf16_t* dst, int K, int N, int tr) {
    long total = (long)K * N;
    convert_w_k<<<dim3((unsigned)((total + 255) / 256)), 256, 0, stream>>>(src, dst, K, N, tr);
  };
  cw(conv_w, wconvT, 768, 768, 0);  // conv_w already [out][in]
  for (int l = 0; l < 4; ++l) {
    cw(blk(l, 2),  wqkvT[l],  768, 2304, 1);
    cw(blk(l, 4),  wprojT[l], 768, 768, 1);
    cw(blk(l, 8),  w1T[l],    768, 3072, 1);
    cw(blk(l, 10), w2T[l],    3072, 768, 1);
  }
  for (int l = 0; l < 3; ++l) cw(spl(l, 0), wsplT[l], 192, 768, 1);

  schedule_k<<<1, 1, 0, stream>>>(ol0, ol1, ol2, meta, scaleA, idxA, scaleB, idxB,
                                  map_src, ssrc, fscale, fidx, Npad);

  patch_embed_k<<<dim3(16, 12, BATCH), 128, 0, stream>>>(im, wconvT, conv_b, pos, X0, Npad);

  const unsigned mt = (unsigned)(Npad / 16);
  float* cur = X0; float* nxt = X1;
  for (int i = 0; i < 4; ++i) {
    const int* cnt = meta + i;
    layernorm_k<<<dim3(Npad / 8, 1, BATCH), 256, 0, stream>>>(cur, blk(i, 0), blk(i, 1),
                                                              Hbf, cnt, NPD);
    gemm16_k<0><<<dim3(mt, 18, BATCH), 128, 0, stream>>>(
        Hbf, wqkvT[i], blk(i, 3), nullptr, QKV, nullptr, cnt, nullptr,
        768, NPD, NPQ, 0, 768, 2304);
    attention_k<<<dim3(mt, N_HEADS, BATCH), 32, 0, stream>>>(QKV, Obf, cnt, Npad);
    gemm16_k<2><<<dim3(mt, 6, BATCH), 128, 0, stream>>>(
        Obf, wprojT[i], blk(i, 5), cur, nullptr, cur, cnt, nullptr,
        768, NPD, NPD, NPD, 768, 768);
    layernorm_k<<<dim3(Npad / 8, 1, BATCH), 256, 0, stream>>>(cur, blk(i, 6), blk(i, 7),
                                                              Hbf, cnt, NPD);
    gemm16_k<1><<<dim3(mt, 24, BATCH), 128, 0, stream>>>(
        Hbf, w1T[i], blk(i, 9), nullptr, Gbf, nullptr, cnt, nullptr,
        768, NPD, NPF, 0, 768, 3072);
    gemm16_k<2><<<dim3(mt, 6, BATCH), 128, 0, stream>>>(
        Gbf, w2T[i], blk(i, 11), cur, nullptr, cur, cnt, nullptr,
        3072, NPF, NPD, NPD, 3072, 768);
    if (i < 3) {
      copy_oldkeep_k<<<dim3(Npad, 1, BATCH), 256, 0, stream>>>(cur, nxt, map_src + i * Npad,
                                                               meta + 4 + i, Npad);
      gather_split_k<<<dim3(Npad, 1, BATCH), 64, 0, stream>>>(cur, Sg, ssrc + i * Npad,
                                                              meta + 8 + i, Npad);
      gemm16_k<3><<<dim3(mt, 6, BATCH), 128, 0, stream>>>(
          Sg, wsplT[i], spl(i, 1), nxt, nullptr, nullptr, meta + 8 + i, meta + 4 + i,
          192, (long)Npad * 192, NPD, 0, 192, 768);
      float* t = cur; cur = nxt; nxt = t;
    }
  }
  write_out_k<<<dim3(Nf, 1, BATCH), 256, 0, stream>>>(cur, fscale, fidx, (float*)d_out,
                                                      Nf, Npad);
}
